// EnhancedSparseMoE_15281493639600
// MI455X (gfx1250) — compile-verified, matrix operand order checked by reading, not run on hardware
//
#include <hip/hip_runtime.h>
#include <hip/hip_bf16.h>
#include <math.h>

// ---------------- problem constants (from reference) ----------------
#define H_DIM 2048   // hidden size
#define I_DIM 1408   // intermediate size
#define E_NUM 8      // experts
#define T_TOK 2048   // B*S tokens
#define MT    32     // tokens per FFN tile
#define KSTEP 32     // WMMA K per step (bf16)

typedef __attribute__((ext_vector_type(16))) __bf16 v16bf;
typedef __attribute__((ext_vector_type(8)))  __bf16 v8bf;
typedef __attribute__((ext_vector_type(8)))  float  v8f;

// ---------------- WMMA helpers ----------------
__device__ __forceinline__ v8f wmma_bf16(v16bf a, v16bf b, v8f c) {
    // v_wmma_f32_16x16x32_bf16: (neg_a, A, neg_b, B, c_mod, C, reuse_a, reuse_b)
    return __builtin_amdgcn_wmma_f32_16x16x32_bf16(false, a, false, b, (short)0, c,
                                                   false, false);
}

__device__ __forceinline__ float silu_fast(float g) {
    // g * sigmoid(g) with v_rcp_f32 instead of IEEE divide
    return g * __builtin_amdgcn_rcpf(1.f + __expf(-g));
}

// A-fragment (16x32 bf16, MxK) from row-major bf16 LDS tile.
// ISA layout: lanes 0-15 hold row M=lane, K = {0..7, 16..23};
//             lanes 16-31 hold row M=lane-16, K = {8..15, 24..31}.
__device__ __forceinline__ v16bf load_frag_lds(const __bf16* p, int row0, int k0,
                                               int stride, int ln, int lh) {
    const __bf16* q = p + (size_t)(row0 + ln) * stride + k0 + 8 * lh;
    v8bf c0 = *(const v8bf*)(q);
    v8bf c1 = *(const v8bf*)(q + 16);
    v16bf r;
#pragma unroll
    for (int i = 0; i < 8; ++i) { r[i] = c0[i]; r[8 + i] = c1[i]; }
    return r;
}

// B-fragment (32x16, KxN) from weights stored row-major [N][K] (torch [out,in]):
// per-lane column N = lane&15 -> contiguous K chunks. fp32 source (convert path).
__device__ __forceinline__ v16bf load_bfrag_f32(const float* rowptr, int k0, int lh) {
    const float* q = rowptr + k0 + 8 * lh;
    v8f f0 = *(const v8f*)(q);
    v8f f1 = *(const v8f*)(q + 16);
    v16bf r;
#pragma unroll
    for (int i = 0; i < 8; ++i) { r[i] = (__bf16)f0[i]; r[8 + i] = (__bf16)f1[i]; }
    return r;
}

// Same, from pre-converted bf16 weights (two 16B loads, no VALU converts).
__device__ __forceinline__ v16bf load_bfrag_bf16(const __bf16* rowptr, int k0, int lh) {
    const __bf16* q = rowptr + k0 + 8 * lh;
    v8bf c0 = *(const v8bf*)(q);
    v8bf c1 = *(const v8bf*)(q + 16);
    v16bf r;
#pragma unroll
    for (int i = 0; i < 8; ++i) { r[i] = c0[i]; r[8 + i] = c1[i]; }
    return r;
}

// ---------------- weight pre-conversion fp32 -> bf16 ----------------
__global__ __launch_bounds__(256) void cvt_f32_bf16_kernel(
    const float* __restrict__ src, __bf16* __restrict__ dst)
{
    const size_t i = ((size_t)blockIdx.x * 256 + threadIdx.x) * 8;
    const float4 a = *(const float4*)(src + i);
    const float4 b = *(const float4*)(src + i + 4);
    v8bf o;
    o[0] = (__bf16)a.x; o[1] = (__bf16)a.y; o[2] = (__bf16)a.z; o[3] = (__bf16)a.w;
    o[4] = (__bf16)b.x; o[5] = (__bf16)b.y; o[6] = (__bf16)b.z; o[7] = (__bf16)b.w;
    *(v8bf*)(dst + i) = o;
}

// ---------------- gating: softmax -> top2 -> renorm -> slot lists ----------------
__global__ __launch_bounds__(256) void moe_gate_kernel(
    const float* __restrict__ x, const float* __restrict__ gw,
    int* __restrict__ counts, int* __restrict__ slot_tok, float* __restrict__ slot_w)
{
    const int wave = threadIdx.x >> 5;
    const int lane = threadIdx.x & 31;
    const int t = blockIdx.x * 8 + wave;
    if (t >= T_TOK) return;

    const float* xr = x + (size_t)t * H_DIM;
    float acc[E_NUM];
#pragma unroll
    for (int e = 0; e < E_NUM; ++e) acc[e] = 0.f;
    for (int h = lane; h < H_DIM; h += 32) {
        const float xv = xr[h];
#pragma unroll
        for (int e = 0; e < E_NUM; ++e) acc[e] += xv * gw[e * H_DIM + h];
    }
#pragma unroll
    for (int e = 0; e < E_NUM; ++e) {
#pragma unroll
        for (int off = 16; off > 0; off >>= 1)
            acc[e] += __shfl_xor(acc[e], off, 32);
    }
    if (lane == 0) {
        float mx = acc[0];
#pragma unroll
        for (int e = 1; e < E_NUM; ++e) mx = fmaxf(mx, acc[e]);
        float p[E_NUM], s = 0.f;
#pragma unroll
        for (int e = 0; e < E_NUM; ++e) { p[e] = __expf(acc[e] - mx); s += p[e]; }
        const float inv = __builtin_amdgcn_rcpf(s);
#pragma unroll
        for (int e = 0; e < E_NUM; ++e) p[e] *= inv;
        int e0 = 0; float p0 = p[0];
#pragma unroll
        for (int e = 1; e < E_NUM; ++e) if (p[e] > p0) { p0 = p[e]; e0 = e; }
        int e1 = (e0 == 0) ? 1 : 0; float p1 = p[e1];
#pragma unroll
        for (int e = 0; e < E_NUM; ++e) if (e != e0 && p[e] > p1) { p1 = p[e]; e1 = e; }
        const float rs = 1.f / (p0 + p1);   // once per token, precise
        const int s0 = atomicAdd(&counts[e0], 1);
        slot_tok[e0 * T_TOK + s0] = t; slot_w[e0 * T_TOK + s0] = p0 * rs;
        const int s1 = atomicAdd(&counts[e1], 1);
        slot_tok[e1 * T_TOK + s1] = t; slot_w[e1 * T_TOK + s1] = p1 * rs;
    }
}

// ---------------- fused expert FFN: (x@w1^T) -> silu*mul -> @w2^T -> combine ----------------
template <bool PRE>
__global__ __launch_bounds__(256) void moe_ffn_kernel(
    const float* __restrict__ x,
    const float* __restrict__ w1f, const float* __restrict__ w2f,
    const __bf16* __restrict__ w1b, const __bf16* __restrict__ w2b,
    const int* __restrict__ counts, const int* __restrict__ slot_tok,
    const float* __restrict__ slot_w, float* __restrict__ out)
{
    __shared__ __align__(16) __bf16 xa[MT * H_DIM];   // 128 KB: gathered x tile (bf16)
    __shared__ __align__(16) __bf16 act[MT * I_DIM];  //  88 KB: silu(g)*u tile (bf16)
    __shared__ int   tok_s[MT];
    __shared__ float w_s[MT];

    const int e    = blockIdx.y;
    const int cnt  = counts[e];
    const int base = blockIdx.x * MT;
    if (base >= cnt) return;  // uniform early-exit for empty tiles

    const int tid = threadIdx.x;
    if (tid < MT) {
        const int row = base + tid;
        int tk; float wt;
        if (row < cnt) { tk = slot_tok[e * T_TOK + row]; wt = slot_w[e * T_TOK + row]; }
        else           { tk = slot_tok[e * T_TOK + base]; wt = 0.f; }  // padded row
        tok_s[tid] = tk; w_s[tid] = wt;
    }
    __syncthreads();

    // Stage gathered x rows into LDS as bf16 (row-major [MT][H]).
    for (int i = tid * 4; i < MT * H_DIM; i += 256 * 4) {
        const int m = i / H_DIM, c = i % H_DIM;   // 4-chunk stays inside one row
        const float4 f = *(const float4*)(x + (size_t)tok_s[m] * H_DIM + c);
        xa[i + 0] = (__bf16)f.x; xa[i + 1] = (__bf16)f.y;
        xa[i + 2] = (__bf16)f.z; xa[i + 3] = (__bf16)f.w;
    }
    __syncthreads();

    const int wave = tid >> 5, lane = tid & 31;
    const int ln = lane & 15, lh = lane >> 4;

    // ---- phase 1: gate_up GEMM + SiluAndMul -> act (LDS) ----
    // Per k-step: 2 A frags reused by gate and up B frags -> 4 WMMA / 8 loads.
    const size_t w1off = (size_t)e * (2 * I_DIM) * H_DIM;   // [2I, H] row-major
    for (int nt = wave; nt < I_DIM / 16; nt += 8) {          // 88 N-tiles / 8 waves
        v8f cg0 = {}, cg1 = {}, cu0 = {}, cu1 = {};
        const size_t gro = w1off + (size_t)(nt * 16 + ln) * H_DIM;           // gate row
        const size_t uro = w1off + (size_t)(I_DIM + nt * 16 + ln) * H_DIM;   // up row
        for (int k0 = 0; k0 < H_DIM; k0 += KSTEP) {          // 64 k-steps
            const v16bf a0 = load_frag_lds(xa, 0,  k0, H_DIM, ln, lh);
            const v16bf a1 = load_frag_lds(xa, 16, k0, H_DIM, ln, lh);
            const v16bf Bg = PRE ? load_bfrag_bf16(w1b + gro, k0, lh)
                                 : load_bfrag_f32 (w1f + gro, k0, lh);
            const v16bf Bu = PRE ? load_bfrag_bf16(w1b + uro, k0, lh)
                                 : load_bfrag_f32 (w1f + uro, k0, lh);
            cg0 = wmma_bf16(a0, Bg, cg0);
            cg1 = wmma_bf16(a1, Bg, cg1);
            cu0 = wmma_bf16(a0, Bu, cu0);
            cu1 = wmma_bf16(a1, Bu, cu1);
        }
        // C/D layout: N = lane&15, M = vgpr + 8*lane[4]; act row-major [MT][I]
#pragma unroll
        for (int v = 0; v < 8; ++v) {
            const int m0 = v + 8 * lh;
            act[(size_t)m0 * I_DIM + nt * 16 + ln] = (__bf16)(silu_fast(cg0[v]) * cu0[v]);
            const int m1 = 16 + v + 8 * lh;
            act[(size_t)m1 * I_DIM + nt * 16 + ln] = (__bf16)(silu_fast(cg1[v]) * cu1[v]);
        }
    }
    __syncthreads();

    // ---- phase 2: down-proj GEMM + weighted scatter-add into out ----
    // 2-way N-blocking: 2 A frags x 2 B frags -> 4 WMMA / 8 loads per k-step.
    const size_t w2off = (size_t)e * H_DIM * I_DIM;          // [H, I] row-major
    for (int np = wave; np < H_DIM / 32; np += 8) {          // 64 pairs / 8 waves = 8
        const int ntA = np * 2;                              // two adjacent N-tiles
        v8f c00 = {}, c01 = {}, c10 = {}, c11 = {};
        const size_t dro0 = w2off + (size_t)(ntA * 16 + ln) * I_DIM;
        const size_t dro1 = w2off + (size_t)(ntA * 16 + 16 + ln) * I_DIM;
        for (int k0 = 0; k0 < I_DIM; k0 += KSTEP) {          // 44 k-steps
            const v16bf a0 = load_frag_lds(act, 0,  k0, I_DIM, ln, lh);
            const v16bf a1 = load_frag_lds(act, 16, k0, I_DIM, ln, lh);
            const v16bf B0 = PRE ? load_bfrag_bf16(w2b + dro0, k0, lh)
                                 : load_bfrag_f32 (w2f + dro0, k0, lh);
            const v16bf B1 = PRE ? load_bfrag_bf16(w2b + dro1, k0, lh)
                                 : load_bfrag_f32 (w2f + dro1, k0, lh);
            c00 = wmma_bf16(a0, B0, c00);
            c10 = wmma_bf16(a1, B0, c10);
            c01 = wmma_bf16(a0, B1, c01);
            c11 = wmma_bf16(a1, B1, c11);
        }
        const int col0 = ntA * 16 + ln;
        const int col1 = col0 + 16;
#pragma unroll
        for (int v = 0; v < 8; ++v) {
            const int m0 = v + 8 * lh;
            const int m1 = 16 + v + 8 * lh;
            const size_t r0 = (size_t)tok_s[m0] * H_DIM;
            const size_t r1 = (size_t)tok_s[m1] * H_DIM;
            atomicAdd(&out[r0 + col0], w_s[m0] * c00[v]);
            atomicAdd(&out[r1 + col0], w_s[m1] * c10[v]);
            atomicAdd(&out[r0 + col1], w_s[m0] * c01[v]);
            atomicAdd(&out[r1 + col1], w_s[m1] * c11[v]);
        }
    }
}

// ---------------- host launcher ----------------
extern "C" void kernel_launch(void* const* d_in, const int* in_sizes, int n_in,
                              void* d_out, int out_size, void* d_ws, size_t ws_size,
                              hipStream_t stream) {
    const float* x  = (const float*)d_in[0];  // [T, H]
    const float* gw = (const float*)d_in[1];  // [E, H]
    const float* w1 = (const float*)d_in[2];  // [E, 2I, H]
    const float* w2 = (const float*)d_in[3];  // [E, H, I]
    float* out = (float*)d_out;

    const size_t n_w1 = (size_t)E_NUM * 2 * I_DIM * H_DIM;  // 46,137,344
    const size_t n_w2 = (size_t)E_NUM * H_DIM * I_DIM;      // 23,068,672

    char* ws = (char*)d_ws;
    size_t off = 0;
    int*   counts   = (int*)(ws + off);  off += 256;
    int*   slot_tok = (int*)(ws + off);  off += (size_t)E_NUM * T_TOK * 4;
    float* slot_w   = (float*)(ws + off); off += (size_t)E_NUM * T_TOK * 4;
    __bf16* w1b = (__bf16*)(ws + off);   off += n_w1 * 2;
    __bf16* w2b = (__bf16*)(ws + off);   off += n_w2 * 2;
    const bool pre = (ws_size >= off);   // constant across calls -> deterministic

    hipMemsetAsync(counts, 0, E_NUM * sizeof(int), stream);
    hipMemsetAsync(out, 0, (size_t)out_size * sizeof(float), stream);

    moe_gate_kernel<<<T_TOK / 8, 256, 0, stream>>>(x, gw, counts, slot_tok, slot_w);

    dim3 grid(T_TOK / MT, E_NUM);  // token tiles fastest -> expert-major L2 locality
    if (pre) {
        cvt_f32_bf16_kernel<<<(int)(n_w1 / 2048), 256, 0, stream>>>(w1, w1b);
        cvt_f32_bf16_kernel<<<(int)(n_w2 / 2048), 256, 0, stream>>>(w2, w2b);
        moe_ffn_kernel<true><<<grid, 256, 0, stream>>>(x, w1, w2, w1b, w2b,
                                                       counts, slot_tok, slot_w, out);
    } else {
        moe_ffn_kernel<false><<<grid, 256, 0, stream>>>(x, w1, w2, w1b, w2b,
                                                        counts, slot_tok, slot_w, out);
    }
}
